// Lumiere1M_12567074308415
// MI455X (gfx1250) — matrix operand order repaired and checked by reading, not compiled, not run on hardware
//
#include <hip/hip_runtime.h>
#include <hip/hip_bf16.h>
#include <math.h>

// ---------------------------------------------------------------------------
// Model dims (fixed by the reference)
// ---------------------------------------------------------------------------
static constexpr int Bb = 2, Ss = 1024, Dd = 768, Hh = 12, DKk = 64, DVv = 64;
static constexpr int DFF = 3072, Vv = 32000, Ll = 4;
static constexpr int BS = Bb * Ss;   // 2048 token rows

typedef _Float16 half_t;
typedef __attribute__((ext_vector_type(16))) _Float16 v16h;
typedef __attribute__((ext_vector_type(8)))  _Float16 v8h;
typedef __attribute__((ext_vector_type(8)))  float    v8f;

// LDS B-tile: stored transposed [n][k], row pitch 72 halves (144 B):
//  - every row start is 16B aligned (144 = 9*16) for b128 LDS loads
//  - lane n reads at n*36 dwords -> 16 distinct banks across a half-wave
static constexpr int BT_PITCH = 72;

// ---------------------------------------------------------------------------
// fp32 -> f16 conversion (weights, once per launch)
// ---------------------------------------------------------------------------
__global__ __launch_bounds__(256)
void cast_f16_kernel(const float* __restrict__ in, half_t* __restrict__ out, long n) {
    long i = (long)blockIdx.x * 256 + threadIdx.x;
    long stride = (long)gridDim.x * 256;
    for (; i < n; i += stride) out[i] = (half_t)in[i];
}

// ---------------------------------------------------------------------------
// Embedding lookup + sinusoidal positional encoding -> h (f32) and hb (f16)
// ---------------------------------------------------------------------------
__global__ __launch_bounds__(256)
void embed_pe_kernel(const int* __restrict__ x, const float* __restrict__ emb,
                     float* __restrict__ h, half_t* __restrict__ hb) {
    int row = blockIdx.x;            // 0..BS-1
    int s   = row % Ss;
    int tok = x[row];
    for (int d = threadIdx.x; d < Dd; d += blockDim.x) {
        float v = emb[(long)tok * Dd + d];
        int   i2    = d >> 1;
        float scale = powf(10000.0f, (2.0f * (float)i2) / (float)Dd);
        float ang   = (float)s / scale;
        v += (d & 1) ? cosf(ang) : sinf(ang);
        h [(long)row * Dd + d] = v;
        hb[(long)row * Dd + d] = (half_t)v;
    }
}

// ---------------------------------------------------------------------------
// Generic batched WMMA GEMM:  C = A * B (+bias) (+gelu)
//   A: [M,K] f16 row-major (lda), B: [K,N] f16 row-major (ldb) or B^T if transB
//   C: f32 (Cf) and/or f16 (Ch), ldc, per-z offsets decomposed as z=(b*Hdim+h)
//   Block = 128 threads = 4 waves; block -> 64x64 C tile (wave: 16 rows x 64).
//   Dense-B path: double-buffered LDS slabs (32x64, transposed+padded) with
//   next-slab global loads issued before the current slab's WMMAs.
//   All launched shapes have M,N multiples of 64 and K multiple of 32.
// ---------------------------------------------------------------------------
__global__ __launch_bounds__(128)
void gemm_wmma_f16_kernel(const half_t* __restrict__ A, const half_t* __restrict__ Bm,
                          const float* __restrict__ bias,
                          float* __restrict__ Cf, half_t* __restrict__ Ch,
                          int M, int N, int K, int lda, int ldb, int ldc,
                          int transB, int gelu, int Hdim,
                          long sAb, long sAh, long sBb, long sBh, long sCb, long sCh) {
    __shared__ half_t bt[2][64 * BT_PITCH];   // 2 x 9216 B

    int z  = blockIdx.z;
    int bb = z / Hdim, hh = z % Hdim;
    A  += bb * sAb + hh * sAh;
    Bm += bb * sBb + hh * sBh;
    long cOff = bb * sCb + hh * sCh;

    int wave = threadIdx.x >> 5;
    int lane = threadIdx.x & 31;
    int lrow = lane & 15;
    int hi   = lane >> 4;              // 0 or 1: which half of the lane group

    int m0 = blockIdx.y * 64 + wave * 16;
    int n0 = blockIdx.x * 64;

    v8f acc[4];
    #pragma unroll
    for (int t = 0; t < 4; ++t) acc[t] = (v8f){0.f,0.f,0.f,0.f,0.f,0.f,0.f,0.f};

    // Per-lane A row base: fragment = 2 contiguous 16B chunks per k-step
    const half_t* Arow = A + (long)(m0 + lrow) * lda + 8 * hi;

    if (transB) {
        // ---- direct path (Q.K^T): per-lane B reads are contiguous along k ----
        const half_t* Brow0 = Bm + (long)(n0 + 0 * 16 + lrow) * ldb + 8 * hi;
        const half_t* Brow1 = Bm + (long)(n0 + 1 * 16 + lrow) * ldb + 8 * hi;
        const half_t* Brow2 = Bm + (long)(n0 + 2 * 16 + lrow) * ldb + 8 * hi;
        const half_t* Brow3 = Bm + (long)(n0 + 3 * 16 + lrow) * ldb + 8 * hi;
        for (int k0 = 0; k0 < K; k0 += 32) {
            v8h a0 = *(const v8h*)(Arow + k0);
            v8h a1 = *(const v8h*)(Arow + k0 + 16);
            v16h af;
            #pragma unroll
            for (int i = 0; i < 8; ++i) { af[i] = a0[i]; af[8 + i] = a1[i]; }
            const half_t* brows[4] = {Brow0, Brow1, Brow2, Brow3};
            #pragma unroll
            for (int t = 0; t < 4; ++t) {
                v8h b0 = *(const v8h*)(brows[t] + k0);
                v8h b1 = *(const v8h*)(brows[t] + k0 + 16);
                v16h bf;
                #pragma unroll
                for (int i = 0; i < 8; ++i) { bf[i] = b0[i]; bf[8 + i] = b1[i]; }
                acc[t] = __builtin_amdgcn_wmma_f32_16x16x32_f16(
                    false, af, false, bf, (short)0, acc[t], false, false);
            }
        }
    } else {
        // ---- LDS-staged path: double-buffered 32x64 B slabs ----
        // Each thread owns 8 cols x 2 adjacent k-rows of the slab:
        //   loads 2x16B from global, stores 8x b32 (k-pairs contiguous in bt)
        int r2 = (threadIdx.x >> 3) * 2;     // 0,2,..,30 : even k row
        int c8 = (threadIdx.x & 7) * 8;      // 0..56     : col group
        const half_t* Bsrc = Bm + (long)r2 * ldb + n0 + c8;
        int ldsA = lrow * BT_PITCH + 8 * hi;

        int nIter = K >> 5;
        // prologue: slab 0 -> buffer 0
        v8h x0 = *(const v8h*)(Bsrc);
        v8h x1 = *(const v8h*)(Bsrc + ldb);
        #pragma unroll
        for (int e = 0; e < 8; ++e) {
            unsigned w = (unsigned)__builtin_bit_cast(unsigned short, x0[e]) |
                         ((unsigned)__builtin_bit_cast(unsigned short, x1[e]) << 16);
            *(unsigned*)(&bt[0][(c8 + e) * BT_PITCH + r2]) = w;
        }
        __syncthreads();

        v8h a0 = *(const v8h*)(Arow);
        v8h a1 = *(const v8h*)(Arow + 16);

        for (int it = 0; it < nIter; ++it) {
            int cur = it & 1;
            bool more = (it + 1 < nIter);
            v8h nx0, nx1, na0, na1;
            if (more) {
                // issue next-slab loads before this slab's WMMAs
                Bsrc += 32 * (long)ldb;
                nx0 = *(const v8h*)(Bsrc);
                nx1 = *(const v8h*)(Bsrc + ldb);
                long kn = (long)(it + 1) * 32;
                na0 = *(const v8h*)(Arow + kn);
                na1 = *(const v8h*)(Arow + kn + 16);
                if (it + 2 < nIter)
                    __builtin_prefetch(Bsrc + 32 * (long)ldb, 0, 1);
            }
            v16h af;
            #pragma unroll
            for (int i = 0; i < 8; ++i) { af[i] = a0[i]; af[8 + i] = a1[i]; }
            #pragma unroll
            for (int t = 0; t < 4; ++t) {
                const half_t* bl = &bt[cur][ldsA + t * 16 * BT_PITCH];
                v8h b0 = *(const v8h*)(bl);
                v8h b1 = *(const v8h*)(bl + 16);
                v16h bf;
                #pragma unroll
                for (int i = 0; i < 8; ++i) { bf[i] = b0[i]; bf[8 + i] = b1[i]; }
                acc[t] = __builtin_amdgcn_wmma_f32_16x16x32_f16(
                    false, af, false, bf, (short)0, acc[t], false, false);
            }
            if (more) {
                #pragma unroll
                for (int e = 0; e < 8; ++e) {
                    unsigned w = (unsigned)__builtin_bit_cast(unsigned short, nx0[e]) |
                                 ((unsigned)__builtin_bit_cast(unsigned short, nx1[e]) << 16);
                    *(unsigned*)(&bt[cur ^ 1][(c8 + e) * BT_PITCH + r2]) = w;
                }
                __syncthreads();
                a0 = na0; a1 = na1;
            }
        }
    }

    // ---- epilogue: bias / exact GELU / store f32 and-or f16 ----
    #pragma unroll
    for (int t = 0; t < 4; ++t) {
        #pragma unroll
        for (int rr = 0; rr < 8; ++rr) {
            int row = m0 + rr + 8 * hi;
            int col = n0 + t * 16 + lrow;
            if (row < M && col < N) {
                float v = acc[t][rr];
                if (bias) v += bias[col];
                if (gelu) v = 0.5f * v * (1.0f + erff(v * 0.70710678118f));
                long idx = cOff + (long)row * ldc + col;
                if (Cf) Cf[idx] = v;
                if (Ch) Ch[idx] = (half_t)v;
            }
        }
    }
}

// ---------------------------------------------------------------------------
// Causal masked "softmax" (reference's exp/(sum+1e-9), no max-subtraction).
// Reads one f32 score row, writes f16 probabilities IN PLACE over the same row
// (row byte-pitch stays Ss*4; as f16 the row pitch is 2*Ss elements).
// Mask: j >= i is masked (triu with diagonal).
// ---------------------------------------------------------------------------
__global__ __launch_bounds__(256)
void softmax_causal_kernel(float* __restrict__ scores, float scale) {
    long row = blockIdx.x;               // over Bb*Hh*Ss rows
    int  i   = (int)(row % Ss);          // query index
    float*  srow = scores + row * (long)Ss;
    half_t* prow = (half_t*)srow;        // aliases the same bytes

    float e[4];
    float sum = 0.f;
    #pragma unroll
    for (int c = 0; c < 4; ++c) {
        int j = threadIdx.x + c * 256;
        float v = (j < i) ? expf(srow[j] * scale) : 0.0f;
        e[c] = v; sum += v;
    }
    __shared__ float red[8];
    for (int o = 16; o > 0; o >>= 1) sum += __shfl_down(sum, o);
    if ((threadIdx.x & 31) == 0) red[threadIdx.x >> 5] = sum;
    __syncthreads();
    __shared__ float inv_s;
    if (threadIdx.x == 0) {
        float ts = 0.f;
        for (int w = 0; w < 8; ++w) ts += red[w];
        inv_s = 1.0f / (ts + 1e-9f);
    }
    __syncthreads();
    float inv = inv_s;
    #pragma unroll
    for (int c = 0; c < 4; ++c) {
        int j = threadIdx.x + c * 256;
        prow[j] = (half_t)(e[c] * inv);
    }
}

// ---------------------------------------------------------------------------
// Fused residual add + LayerNorm -> h (f32) and hb (f16). One row per block.
// ---------------------------------------------------------------------------
__global__ __launch_bounds__(256)
void add_ln_kernel(float* __restrict__ h, const float* __restrict__ delta,
                   const float* __restrict__ g, const float* __restrict__ be,
                   half_t* __restrict__ hb) {
    int row = blockIdx.x;
    float*       hr = h     + (long)row * Dd;
    const float* dr = delta + (long)row * Dd;
    float vals[3]; float s = 0.f, sq = 0.f;
    #pragma unroll
    for (int c = 0; c < 3; ++c) {
        int d = threadIdx.x + c * 256;
        float v = hr[d] + dr[d];
        vals[c] = v; s += v; sq += v * v;
    }
    __shared__ float r1[8], r2[8];
    for (int o = 16; o > 0; o >>= 1) { s += __shfl_down(s, o); sq += __shfl_down(sq, o); }
    if ((threadIdx.x & 31) == 0) { r1[threadIdx.x >> 5] = s; r2[threadIdx.x >> 5] = sq; }
    __syncthreads();
    __shared__ float mean_s, rstd_s;
    if (threadIdx.x == 0) {
        float ts = 0.f, tq = 0.f;
        for (int w = 0; w < 8; ++w) { ts += r1[w]; tq += r2[w]; }
        float mean = ts / (float)Dd;
        float var  = tq / (float)Dd - mean * mean;
        mean_s = mean; rstd_s = rsqrtf(var + 1e-5f);
    }
    __syncthreads();
    float mean = mean_s, rstd = rstd_s;
    #pragma unroll
    for (int c = 0; c < 3; ++c) {
        int d = threadIdx.x + c * 256;
        float v = (vals[c] - mean) * rstd * g[d] + be[d];
        hr[d] = v;
        hb[(long)row * Dd + d] = (half_t)v;
    }
}

// ---------------------------------------------------------------------------
// Host side
// ---------------------------------------------------------------------------
static inline void run_cast(const float* in, half_t* out, long n, hipStream_t st) {
    int blocks = (int)((n + 255) / 256);
    if (blocks > 8192) blocks = 8192;
    cast_f16_kernel<<<blocks, 256, 0, st>>>(in, out, n);
}

static inline void run_gemm(const half_t* A, const half_t* Bm, const float* bias,
                            float* Cf, half_t* Ch,
                            int M, int N, int K, int lda, int ldb, int ldc,
                            int transB, int gelu, int Z, int Hdim,
                            long sAb, long sAh, long sBb, long sBh, long sCb, long sCh,
                            hipStream_t st) {
    dim3 grid((N + 63) / 64, (M + 63) / 64, Z);
    gemm_wmma_f16_kernel<<<grid, 128, 0, st>>>(A, Bm, bias, Cf, Ch, M, N, K,
                                               lda, ldb, ldc, transB, gelu, Hdim,
                                               sAb, sAh, sBb, sBh, sCb, sCh);
}

extern "C" void kernel_launch(void* const* d_in, const int* in_sizes, int n_in,
                              void* d_out, int out_size, void* d_ws, size_t ws_size,
                              hipStream_t stream) {
    (void)in_sizes; (void)n_in; (void)out_size; (void)ws_size;

    const int*   x_ids = (const int*)  d_in[0];
    const float* emb   = (const float*)d_in[1];
    const float* Wq    = (const float*)d_in[2];
    const float* Wk    = (const float*)d_in[3];
    const float* Wv    = (const float*)d_in[4];
    const float* Wo    = (const float*)d_in[5];
    const float* ln1_s = (const float*)d_in[6];
    const float* ln1_b = (const float*)d_in[7];
    const float* W1    = (const float*)d_in[8];
    const float* b1    = (const float*)d_in[9];
    const float* W2    = (const float*)d_in[10];
    const float* b2    = (const float*)d_in[11];
    const float* ln2_s = (const float*)d_in[12];
    const float* ln2_b = (const float*)d_in[13];
    const float* lm_W  = (const float*)d_in[14];
    const float* lm_b  = (const float*)d_in[15];
    float* out = (float*)d_out;

    // ---- workspace carve-out (256B aligned) ----
    char* base = (char*)d_ws;
    size_t off = 0;
    auto carve = [&](size_t bytes) -> char* {
        char* p = base + off;
        off = (off + bytes + 255) & ~(size_t)255;
        return p;
    };
    float*  h_f    = (float*) carve((size_t)BS * Dd * 4);
    half_t* hb     = (half_t*)carve((size_t)BS * Dd * 2);
    half_t* wq_h   = (half_t*)carve((size_t)Ll * Dd * Dd * 2);
    half_t* wk_h   = (half_t*)carve((size_t)Ll * Dd * Dd * 2);
    half_t* wv_h   = (half_t*)carve((size_t)Ll * Dd * Dd * 2);
    half_t* wo_h   = (half_t*)carve((size_t)Ll * Dd * Dd * 2);
    half_t* w1_h   = (half_t*)carve((size_t)Ll * Dd * DFF * 2);
    half_t* w2_h   = (half_t*)carve((size_t)Ll * DFF * Dd * 2);
    half_t* lmw_h  = (half_t*)carve((size_t)Dd * Vv * 2);
    half_t* q_h    = (half_t*)carve((size_t)BS * Dd * 2);
    half_t* k_h    = (half_t*)carve((size_t)BS * Dd * 2);
    half_t* v_h    = (half_t*)carve((size_t)BS * Dd * 2);
    float*  scores = (float*) carve((size_t)Bb * Hh * Ss * Ss * 4); // probs alias (f16) in place
    half_t* av_h   = (half_t*)carve((size_t)BS * Dd * 2);
    half_t* ff1_h  = (half_t*)carve((size_t)BS * DFF * 2);
    float*  tmp_f  = (float*) carve((size_t)BS * Dd * 4);

    // ---- one-time weight conversion fp32 -> f16 ----
    run_cast(Wq,   wq_h,  (long)Ll * Dd * Dd,  stream);
    run_cast(Wk,   wk_h,  (long)Ll * Dd * Dd,  stream);
    run_cast(Wv,   wv_h,  (long)Ll * Dd * Dd,  stream);
    run_cast(Wo,   wo_h,  (long)Ll * Dd * Dd,  stream);
    run_cast(W1,   w1_h,  (long)Ll * Dd * DFF, stream);
    run_cast(W2,   w2_h,  (long)Ll * DFF * Dd, stream);
    run_cast(lm_W, lmw_h, (long)Dd * Vv,       stream);

    // ---- embedding + positional encoding ----
    embed_pe_kernel<<<BS, 256, 0, stream>>>(x_ids, emb, h_f, hb);

    const float inv_sqrt_dk = 0.125f;  // 1/sqrt(64)

    for (int l = 0; l < Ll; ++l) {
        const half_t* wq = wq_h + (long)l * Dd * Dd;
        const half_t* wk = wk_h + (long)l * Dd * Dd;
        const half_t* wv = wv_h + (long)l * Dd * Dd;
        const half_t* wo = wo_h + (long)l * Dd * Dd;
        const half_t* w1 = w1_h + (long)l * Dd * DFF;
        const half_t* w2 = w2_h + (long)l * DFF * Dd;

        // Q, K, V projections: [2048,768] x [768,768] -> f16
        run_gemm(hb, wq, nullptr, nullptr, q_h, BS, Dd, Dd, Dd, Dd, Dd,
                 0, 0, 1, 1, 0,0,0,0,0,0, stream);
        run_gemm(hb, wk, nullptr, nullptr, k_h, BS, Dd, Dd, Dd, Dd, Dd,
                 0, 0, 1, 1, 0,0,0,0,0,0, stream);
        run_gemm(hb, wv, nullptr, nullptr, v_h, BS, Dd, Dd, Dd, Dd, Dd,
                 0, 0, 1, 1, 0,0,0,0,0,0, stream);

        // scores = Q . K^T per (b,h): [1024,64] x [64,1024] -> f32
        run_gemm(q_h, k_h, nullptr, scores, nullptr,
                 Ss, Ss, DKk, Dd, Dd, Ss,
                 /*transB=*/1, 0, Bb * Hh, Hh,
                 (long)Ss * Dd, 64, (long)Ss * Dd, 64,
                 (long)Hh * Ss * Ss, (long)Ss * Ss, stream);

        // masked exp/(sum+eps) softmax, f16 probs written in place
        softmax_causal_kernel<<<Bb * Hh * Ss, 256, 0, stream>>>(scores, inv_sqrt_dk);

        // AV = probs . V per (b,h): [1024,1024](f16, pitch 2*Ss) x [1024,64] -> f16
        run_gemm((const half_t*)scores, v_h, nullptr, nullptr, av_h,
                 Ss, DVv, Ss, 2 * Ss, Dd, Dd,
                 0, 0, Bb * Hh, Hh,
                 (long)Hh * Ss * Ss * 2, (long)Ss * Ss * 2,   // A strides in f16 elems
                 (long)Ss * Dd, 64,
                 (long)Ss * Dd, 64, stream);

        // attn out projection: [2048,768] x [768,768] -> f32 tmp
        run_gemm(av_h, wo, nullptr, tmp_f, nullptr, BS, Dd, Dd, Dd, Dd, Dd,
                 0, 0, 1, 1, 0,0,0,0,0,0, stream);

        // h = LN(h + attn)
        add_ln_kernel<<<BS, 256, 0, stream>>>(h_f, tmp_f,
                                              ln1_s + (long)l * Dd, ln1_b + (long)l * Dd, hb);

        // FFN1: [2048,768] x [768,3072] + b1, exact GELU -> f16
        run_gemm(hb, w1, b1 + (long)l * DFF, nullptr, ff1_h,
                 BS, DFF, Dd, Dd, DFF, DFF, 0, /*gelu=*/1, 1, 1,
                 0,0,0,0,0,0, stream);

        // FFN2: [2048,3072] x [3072,768] + b2 -> f32 tmp
        run_gemm(ff1_h, w2, b2 + (long)l * Dd, tmp_f, nullptr,
                 BS, Dd, DFF, DFF, Dd, Dd, 0, 0, 1, 1,
                 0,0,0,0,0,0, stream);

        // h = LN(h + ff)
        add_ln_kernel<<<BS, 256, 0, stream>>>(h_f, tmp_f,
                                              ln2_s + (long)l * Dd, ln2_b + (long)l * Dd, hb);
    }

    // lm head: [2048,768] x [768,32000] + lm_b -> f32 logits
    run_gemm(hb, lmw_h, lm_b, out, nullptr, BS, Vv, Dd, Dd, Vv, Vv,
             0, 0, 1, 1, 0,0,0,0,0,0, stream);
}
